// GCN_11562051961220
// MI455X (gfx1250) — matrix operand order repaired and checked by reading, not compiled, not run on hardware
//
#include <hip/hip_runtime.h>

// ---------------------------------------------------------------------------
// GCN 2-layer forward for MI455X (gfx1250, wave32).
//   layer: out = scatter_add_{A+I}( (x@W) * dinv[src]*dinv[dst] ) + b  [, relu]
// GEMMs: v_wmma_f32_16x16x32_f16, fully unrolled over K (templated K,N).
// Aggregation: hardware global_atomic_add_f32; ~115MB working set is
// L2-resident (192MB), so the scatter phase runs at L2 rate, not HBM.
// dinv[src] is pre-folded into h so the edge loop gathers only dinv[dst].
// ---------------------------------------------------------------------------

typedef __attribute__((ext_vector_type(16))) _Float16 v16h;
typedef __attribute__((ext_vector_type(8)))  float    v8f;

static __device__ __forceinline__ void atomic_add_f32(float* p, float v) {
  unsafeAtomicAdd(p, v);   // -> global_atomic_add_f32 (no CAS loop)
}

// ---- degree / normalization --------------------------------------------------
__global__ void k_deg_init(float* __restrict__ deg, int n) {
  int i = blockIdx.x * blockDim.x + threadIdx.x;
  if (i < n) deg[i] = 1.0f;                       // self-loop contribution
}

__global__ void k_deg_accum(const int* __restrict__ dst, float* __restrict__ deg, int ne) {
  int i = blockIdx.x * blockDim.x + threadIdx.x;
  if (i < ne) atomic_add_f32(&deg[dst[i]], 1.0f);
}

__global__ void k_deg_finalize(float* __restrict__ deg, int n) {
  int i = blockIdx.x * blockDim.x + threadIdx.x;
  if (i < n) deg[i] = rsqrtf(deg[i]);             // deg > 0 always (self-loop)
}

// ---- WMMA GEMM: H[M,N] = A[M,K] @ W[K,N], fp32 in/out, f16 multiply ----------
// One wave per 16x16 output tile. blockIdx.x = M-tile (M must be mult of 16),
// wave id within block = N-tile. K, N compile-time -> K-loop fully unrolled.
template <int K, int N>
__global__ void k_gemm_wmma(const float* __restrict__ A, const float* __restrict__ W,
                            float* __restrict__ H) {
  const int lane  = threadIdx.x & 31;
  const int wave  = threadIdx.x >> 5;
  const int r     = lane & 15;                    // row (A) / col (B,D) in tile
  const int hi    = lane >> 4;                    // half-wave selector
  const int mbase = blockIdx.x << 4;
  const int nbase = wave << 4;

  v8f acc = {};
#pragma unroll
  for (int k0 = 0; k0 < K; k0 += 32) {
    // A fragment (16x32 f16): lane half 'hi' holds K = hi*16 + [0..15]
    v16h a;
    const float* arow = A + (size_t)(mbase + r) * K + (k0 + hi * 16);
#pragma unroll
    for (int t = 0; t < 8; ++t) a[t]     = (_Float16)arow[t];
#pragma unroll
    for (int t = 0; t < 8; ++t) a[8 + t] = (_Float16)arow[8 + t];
    // B fragment (32x16 f16): column nbase+r, K = k0 + hi*16 + [0..15]
    v16h b;
    const float* bcol = W + (size_t)(k0 + hi * 16) * N + (nbase + r);
#pragma unroll
    for (int t = 0; t < 16; ++t) b[t] = (_Float16)bcol[(size_t)t * N];

    acc = __builtin_amdgcn_wmma_f32_16x16x32_f16(
        /*neg_a=*/false, a, /*neg_b=*/false, b,
        /*c_mod=*/(short)0, acc, /*reuse_a=*/false, /*reuse_b=*/false);
  }
  // D layout: VGPR v -> M = hi*8 + v, N = nbase + r
  float* hp = H + (size_t)(mbase + hi * 8) * N + (nbase + r);
#pragma unroll
  for (int v = 0; v < 8; ++v) hp[(size_t)v * N] = acc[v];
}

// ---- fold dinv into h (in place) + emit self-loop term -----------------------
// h[i,f] <- h[i,f]*dinv[i]   (so edge msgs only need dinv[dst])
// agg[i,f] = h_scaled[i,f]*dinv[i]  ( == h*dinv^2, the self-loop term; also
// zero-initializes the accumulator). F = 1<<logF.
__global__ void k_self_scale(float* __restrict__ h, const float* __restrict__ dinv,
                             float* __restrict__ agg, int n, int logF) {
  size_t t = blockIdx.x * (size_t)blockDim.x + threadIdx.x;
  if (t >= ((size_t)n << logF)) return;
  float di = dinv[(int)(t >> logF)];
  float v  = h[t] * di;
  h[t]   = v;
  agg[t] = v * di;
}

// ---- edge scatter: out[dst] += h_scaled[src] * dinv[dst] ---------------------
// One thread per (edge, float4-chunk); chunks = F/4 = 1<<logC.
__global__ void k_edge_scatter(const float* __restrict__ h,
                               const int* __restrict__ src, const int* __restrict__ dst,
                               const float* __restrict__ dinv, float* __restrict__ out,
                               int ne, int logC, int F) {
  size_t t = blockIdx.x * (size_t)blockDim.x + threadIdx.x;
  if (t >= ((size_t)ne << logC)) return;
  int e = (int)(t >> logC);
  int c = (int)(t & ((1u << logC) - 1u));
  int s = src[e];
  int d = dst[e];
  float w = dinv[d];                                      // dinv[src] pre-folded
  float4 m = *((const float4*)(h + (size_t)s * F) + c);   // gather 128b
  float* op = out + (size_t)d * F + (c << 2);
  atomic_add_f32(op + 0, m.x * w);
  atomic_add_f32(op + 1, m.y * w);
  atomic_add_f32(op + 2, m.z * w);
  atomic_add_f32(op + 3, m.w * w);
}

// ---- bias (+ optional relu); may run in place --------------------------------
__global__ void k_bias_act(const float* __restrict__ agg, const float* __restrict__ b,
                           float* __restrict__ out, size_t total, unsigned fmask, int relu) {
  size_t t = blockIdx.x * (size_t)blockDim.x + threadIdx.x;
  if (t >= total) return;
  float v = agg[t] + b[t & fmask];
  out[t] = relu ? fmaxf(v, 0.0f) : v;
}

// ---------------------------------------------------------------------------
extern "C" void kernel_launch(void* const* d_in, const int* in_sizes, int n_in,
                              void* d_out, int out_size, void* d_ws, size_t ws_size,
                              hipStream_t stream) {
  const float* x  = (const float*)d_in[0];   // [n,128]
  const int*   ei = (const int*)  d_in[1];   // [2,e] (JAX default x64-off -> int32)
  const float* W1 = (const float*)d_in[2];   // [128,128]
  const float* b1 = (const float*)d_in[3];   // [128]
  const float* W2 = (const float*)d_in[4];   // [128,64]
  const float* b2 = (const float*)d_in[5];   // [64]

  const int n = in_sizes[0] / 128;           // 100000 (multiple of 16)
  const int e = in_sizes[1] / 2;             // 1600000
  const int* src = ei;
  const int* dst = ei + e;
  float* out = (float*)d_out;                // [n,64]

  // workspace carve-up (f32): dinv | h1[n*128] | agg1[n*128] | h2[n*64]
  float* dinv = (float*)d_ws;
  size_t npad = ((size_t)n + 255) & ~(size_t)255;
  float* h1   = dinv + npad;
  float* agg1 = h1   + (size_t)n * 128;
  float* h2   = agg1 + (size_t)n * 128;
  float* a1   = h1;                          // relu output reuses h1 (h1 dead by then)

  const int B = 256;
  auto blk = [](size_t tot, int b) { return (unsigned)((tot + b - 1) / b); };

  // D^-1/2 of (A + I)
  k_deg_init    <<<blk(n, B), B, 0, stream>>>(dinv, n);
  k_deg_accum   <<<blk(e, B), B, 0, stream>>>(dst, dinv, e);
  k_deg_finalize<<<blk(n, B), B, 0, stream>>>(dinv, n);

  // ---- layer 1: F = 128 ----
  k_gemm_wmma<128, 128><<<n / 16, 256, 0, stream>>>(x, W1, h1);   // 8 waves = 8 N-tiles
  k_self_scale  <<<blk((size_t)n * 128, B), B, 0, stream>>>(h1, dinv, agg1, n, 7);
  k_edge_scatter<<<blk((size_t)e * 32, B), B, 0, stream>>>(h1, src, dst, dinv, agg1, e, 5, 128);
  k_bias_act    <<<blk((size_t)n * 128, B), B, 0, stream>>>(agg1, b1, a1, (size_t)n * 128, 127u, 1);

  // ---- layer 2: F = 64, accumulate straight into d_out ----
  k_gemm_wmma<128, 64><<<n / 16, 128, 0, stream>>>(a1, W2, h2);   // 4 waves = 4 N-tiles
  k_self_scale  <<<blk((size_t)n * 64, B), B, 0, stream>>>(h2, dinv, out, n, 6);
  k_edge_scatter<<<blk((size_t)e * 16, B), B, 0, stream>>>(h2, src, dst, dinv, out, e, 4, 64);
  k_bias_act    <<<blk((size_t)n * 64, B), B, 0, stream>>>(out, b2, out, (size_t)n * 64, 63u, 0);
}
